// MotionEncoder_11235634446365
// MI455X (gfx1250) — matrix operand order repaired and checked by reading, not compiled, not run on hardware
//
#include <hip/hip_runtime.h>

// ---- problem constants (match reference) ----
#define T_DIM 256
#define B_DIM 128
#define D_DIM 1024
#define G3    (3 * D_DIM)
#define N_LAYERS 3

typedef __attribute__((ext_vector_type(16))) __bf16        v16bf;
typedef __attribute__((ext_vector_type(8)))  float         v8f;
typedef __attribute__((ext_vector_type(8)))  unsigned short ushort8;

static __device__ __forceinline__ unsigned short f2bf(float f) {
  unsigned int u = __float_as_uint(f);
  u += 0x7FFFu + ((u >> 16) & 1u);           // round-to-nearest-even
  return (unsigned short)(u >> 16);
}
static __device__ __forceinline__ float bf2f(unsigned short h) {
  return __uint_as_float(((unsigned int)h) << 16);
}
static __device__ __forceinline__ v16bf frag2(ushort8 lo, ushort8 hi) {
  union { ushort8 h[2]; v16bf v; } u;
  u.h[0] = lo; u.h[1] = hi;
  return u.v;
}
static __device__ __forceinline__ v16bf fragld(const unsigned short* p) {
  union { ushort8 h[2]; v16bf v; } u;
  u.h[0] = *(const ushort8*)p;
  u.h[1] = *(const ushort8*)(p + 8);
  return u.v;
}
// Async copy of one 16-byte chunk global -> LDS (tracked by ASYNCcnt).
static __device__ __forceinline__ void async_g2l_b128(unsigned lds_off,
                                                      const void* gaddr) {
  asm volatile("global_load_async_to_lds_b128 %0, %1, off"
               :: "v"(lds_off), "v"(gaddr) : "memory");
}
static __device__ __forceinline__ void wait_async0() {
  asm volatile("s_wait_asynccnt 0x0" ::: "memory");
}

// ---------------- f32 -> bf16 conversion ----------------
__global__ void cvt_f32_bf16(const float* __restrict__ src,
                             unsigned short* __restrict__ dst, int n) {
  int i = (blockIdx.x * blockDim.x + threadIdx.x) * 4;
  if (i + 3 < n) {
    dst[i + 0] = f2bf(src[i + 0]);
    dst[i + 1] = f2bf(src[i + 1]);
    dst[i + 2] = f2bf(src[i + 2]);
    dst[i + 3] = f2bf(src[i + 3]);
  } else {
    for (; i < n; ++i) dst[i] = f2bf(src[i]);
  }
}

// ---------------- zero fill (u32 granularity) ----------------
__global__ void zero_u32(unsigned int* __restrict__ p, int n) {
  int i = blockIdx.x * blockDim.x + threadIdx.x;
  if (i < n) p[i] = 0u;
}

// ---------------- fused dilated GRU step ----------------
// One launch processes `lag` independent timesteps (grid.z) = one round of
// the dilated recurrence.  Block = 8 waves; wave w owns a 32x16 (batch x d)
// output slab (two 16x16 M-tiles sharing all six B weight fragments ->
// halves L2 weight traffic).  A-operands (x_t rows, h_lag rows; 32 rows each)
// are staged into LDS with GLOBAL_LOAD_ASYNC_TO_LDS_B128.
__global__ __launch_bounds__(256)
void gru_step(const unsigned short* __restrict__ in_bf,   // [T,B,D] layer input (bf16)
              unsigned short* __restrict__ out_bf,        // [T,B,D] next-layer input, or null
              float* __restrict__ out_f32,                // [T,B,D] final output, or null
              const unsigned short* __restrict__ Wih,     // [3D,D] bf16 row-major
              const unsigned short* __restrict__ Whh,     // [3D,D] bf16 row-major
              const float* __restrict__ b_ih,             // [3D]
              const float* __restrict__ b_hh,             // [3D]
              float* __restrict__ ring_f,                 // [ringsz,B,D] hidden f32
              unsigned short* __restrict__ ring_b,        // [ringsz,B,D] hidden bf16
              int t0, int lag, int ringsz)
{
  __shared__ unsigned short xa[32 * D_DIM];   // 64 KB: input rows (A operand)
  __shared__ unsigned short ha[32 * D_DIM];   // 64 KB: lagged hidden rows

  const int t       = t0 + blockIdx.z;
  const int slot_w  = t % ringsz;
  const int slot_r  = (t + lag) % ringsz;     // == (t - lag) mod ringsz
  const int rowbase = blockIdx.y * 32;

  // Stage A tiles: 32 consecutive batch rows are contiguous -> linear async copy.
  {
    const unsigned short* sx = in_bf  + (size_t)t      * B_DIM * D_DIM
                                      + (size_t)rowbase * D_DIM;
    const unsigned short* sh = ring_b + (size_t)slot_r * B_DIM * D_DIM
                                      + (size_t)rowbase * D_DIM;
    const unsigned lx = (unsigned)(unsigned long long)&xa[0];
    const unsigned lh = (unsigned)(unsigned long long)&ha[0];
    const int chunks = 32 * D_DIM / 8;        // 4096 x 16B per array
    for (int i = threadIdx.x; i < chunks; i += 256) {
      const unsigned boff = (unsigned)i * 16u;
      async_g2l_b128(lx + boff, sx + i * 8);
      async_g2l_b128(lh + boff, sh + i * 8);
    }
    wait_async0();
  }
  __syncthreads();

  const int wave = threadIdx.x >> 5;
  const int lane = threadIdx.x & 31;
  const int kh   = lane >> 4;                 // lane half (0/1)
  const int l16  = lane & 15;                 // row (A) / col (B,C) index

  const int dcol = blockIdx.x * 128 + wave * 16 + l16;   // column within D

  // Accumulators init = bias (broadcast over rows).
  const float br  = b_ih[dcol]             + b_hh[dcol];
  const float bz  = b_ih[D_DIM + dcol]     + b_hh[D_DIM + dcol];
  const float bin = b_ih[2 * D_DIM + dcol];
  const float bhn = b_hh[2 * D_DIM + dcol];
  v8f acc_r[2], acc_z[2], acc_in[2], acc_hn[2];
  #pragma unroll
  for (int rt = 0; rt < 2; ++rt)
    #pragma unroll
    for (int e = 0; e < 8; ++e) {
      acc_r[rt][e] = br; acc_z[rt][e] = bz;
      acc_in[rt][e] = bin; acc_hn[rt][e] = bhn;
    }

  // B-operand row pointers (W row j = contiguous K) — served from L2.
  const unsigned short* wr_i = Wih + (size_t)dcol * D_DIM;
  const unsigned short* wz_i = Wih + (size_t)(D_DIM + dcol) * D_DIM;
  const unsigned short* wn_i = Wih + (size_t)(2 * D_DIM + dcol) * D_DIM;
  const unsigned short* wr_h = Whh + (size_t)dcol * D_DIM;
  const unsigned short* wz_h = Whh + (size_t)(D_DIM + dcol) * D_DIM;
  const unsigned short* wn_h = Whh + (size_t)(2 * D_DIM + dcol) * D_DIM;

  for (int kt = 0; kt < D_DIM / 32; ++kt) {
    const int k0 = kt * 32;

    // A fragments from LDS (16x32 bf16 layout: lane<16 K=0..15; K=16..31 else)
    v16bf a_x[2], a_h[2];
    #pragma unroll
    for (int rt = 0; rt < 2; ++rt) {
      const unsigned short* ax0 = xa + (rt * 16 + l16) * D_DIM + k0 + kh * 16;
      const unsigned short* ah0 = ha + (rt * 16 + l16) * D_DIM + k0 + kh * 16;
      a_x[rt] = frag2(*(const ushort8*)ax0, *(const ushort8*)(ax0 + 8));
      a_h[rt] = frag2(*(const ushort8*)ah0, *(const ushort8*)(ah0 + 8));
    }

    // B fragments: 16 contiguous K values per lane (32x16 layout)
    const int kb = k0 + kh * 16;
    v16bf b_ir = fragld(wr_i + kb);
    v16bf b_hr = fragld(wr_h + kb);
    v16bf b_iz = fragld(wz_i + kb);
    v16bf b_hz = fragld(wz_h + kb);
    v16bf b_in = fragld(wn_i + kb);
    v16bf b_hn = fragld(wn_h + kb);
    __builtin_prefetch(wr_i + kb + 32, 0, 3);   // global_prefetch_b8: next K tile
    __builtin_prefetch(wr_h + kb + 32, 0, 3);

    // r,z: x- and h-GEMMs share one accumulator (sigmoid(i+h)); n kept split.
    #pragma unroll
    for (int rt = 0; rt < 2; ++rt) {
      acc_r[rt]  = __builtin_amdgcn_wmma_f32_16x16x32_bf16(false, a_x[rt], false, b_ir, (short)0, acc_r[rt],  false, false);
      acc_r[rt]  = __builtin_amdgcn_wmma_f32_16x16x32_bf16(false, a_h[rt], false, b_hr, (short)0, acc_r[rt],  false, false);
      acc_z[rt]  = __builtin_amdgcn_wmma_f32_16x16x32_bf16(false, a_x[rt], false, b_iz, (short)0, acc_z[rt],  false, false);
      acc_z[rt]  = __builtin_amdgcn_wmma_f32_16x16x32_bf16(false, a_h[rt], false, b_hz, (short)0, acc_z[rt],  false, false);
      acc_in[rt] = __builtin_amdgcn_wmma_f32_16x16x32_bf16(false, a_x[rt], false, b_in, (short)0, acc_in[rt], false, false);
      acc_hn[rt] = __builtin_amdgcn_wmma_f32_16x16x32_bf16(false, a_h[rt], false, b_hn, (short)0, acc_hn[rt], false, false);
    }
  }

  // Epilogue: gates, hidden update, residual + LeakyReLU, writeback.
  float*          hw_f = ring_f + (size_t)slot_w * B_DIM * D_DIM;
  unsigned short* hw_b = ring_b + (size_t)slot_w * B_DIM * D_DIM;
  const float*    hr_f = ring_f + (size_t)slot_r * B_DIM * D_DIM;
  const size_t    yoff = (size_t)t * B_DIM * D_DIM;

  #pragma unroll
  for (int rt = 0; rt < 2; ++rt) {
    #pragma unroll
    for (int e = 0; e < 8; ++e) {
      const int M = rt * 16 + e + kh * 8;              // C/D layout: M = vgpr + 8*half
      const size_t idx = (size_t)(rowbase + M) * D_DIM + dcol;
      const float r  = 1.0f / (1.0f + __expf(-acc_r[rt][e]));
      const float z  = 1.0f / (1.0f + __expf(-acc_z[rt][e]));
      const float pre = acc_in[rt][e] + r * acc_hn[rt][e];
      const float ex = __expf(-2.0f * pre);
      const float n  = (1.0f - ex) / (1.0f + ex);      // tanh
      const float hp = hr_f[idx];                       // exact f32 h[t-lag]
      const float hnew = (1.0f - z) * n + z * hp;
      const float hin  = bf2f(xa[M * D_DIM + dcol]);    // layer input (residual)
      float y = (hnew + hin) * 0.70710678118654752f;    // 1/sqrt(2)
      y = y > 0.0f ? y : 0.2f * y;                      // leaky_relu(0.2)
      hw_f[idx] = hnew;
      hw_b[idx] = f2bf(hnew);
      if (out_bf)  out_bf[yoff + idx]  = f2bf(y);
      if (out_f32) out_f32[yoff + idx] = y;
    }
  }
}

// ---------------- host-side orchestration ----------------
extern "C" void kernel_launch(void* const* d_in, const int* in_sizes, int n_in,
                              void* d_out, int out_size, void* d_ws, size_t ws_size,
                              hipStream_t stream) {
  (void)in_sizes; (void)n_in; (void)out_size; (void)ws_size;
  const float* x    = (const float*)d_in[0];   // [1,T,B,D]
  const float* W_ih = (const float*)d_in[1];   // [L,3D,D]
  const float* W_hh = (const float*)d_in[2];   // [L,3D,D]
  const float* b_ih = (const float*)d_in[3];   // [L,3D]
  const float* b_hh = (const float*)d_in[4];   // [L,3D]
  float* out = (float*)d_out;                  // [1,T,B,D]

  // workspace carve-out (all 256B aligned by construction)
  char* ws = (char*)d_ws;
  size_t off = 0;
  auto carve = [&](size_t bytes) -> void* {
    void* p = ws + off;
    off += (bytes + 255) & ~(size_t)255;
    return p;
  };
  const size_t nW   = (size_t)N_LAYERS * G3 * D_DIM;     // 9,437,184
  const size_t nX   = (size_t)T_DIM * B_DIM * D_DIM;     // 33,554,432
  const size_t nRng = (size_t)8 * B_DIM * D_DIM;         // max ring = 2*4 slots

  unsigned short* wih_bf = (unsigned short*)carve(nW * 2);
  unsigned short* whh_bf = (unsigned short*)carve(nW * 2);
  unsigned short* bufA   = (unsigned short*)carve(nX * 2);   // x_bf / layer-1 out
  unsigned short* bufB   = (unsigned short*)carve(nX * 2);   // layer-0 out
  float*          ring_f = (float*)carve(nRng * 4);
  unsigned short* ring_b = (unsigned short*)carve(nRng * 2);

  // 1) convert x and weights to bf16 (weights then live in L2: 38 MB)
  cvt_f32_bf16<<<dim3((unsigned)(nX / 1024)), 256, 0, stream>>>(x, bufA, (int)nX);
  cvt_f32_bf16<<<dim3((unsigned)(nW / 1024)), 256, 0, stream>>>(W_ih, wih_bf, (int)nW);
  cvt_f32_bf16<<<dim3((unsigned)(nW / 1024)), 256, 0, stream>>>(W_hh, whh_bf, (int)nW);

  // 2) layers sequentially over time; per-layer dilated rounds of `lag` steps
  for (int l = 0; l < N_LAYERS; ++l) {
    const int lag    = 1 << l;
    const int ringsz = 2 * lag;

    // zero hidden rings
    zero_u32<<<dim3((unsigned)(nRng / 256)), 256, 0, stream>>>(
        (unsigned int*)ring_f, (int)nRng);
    zero_u32<<<dim3((unsigned)(nRng / 2 / 256)), 256, 0, stream>>>(
        (unsigned int*)ring_b, (int)(nRng / 2));

    const unsigned short* in_bf  = (l == 1) ? bufB : bufA;
    unsigned short*       out_bf = (l == 0) ? bufB : (l == 1 ? bufA : nullptr);
    float*                out_f  = (l == 2) ? out : nullptr;
    const unsigned short* wih_l  = wih_bf + (size_t)l * G3 * D_DIM;
    const unsigned short* whh_l  = whh_bf + (size_t)l * G3 * D_DIM;
    const float*          bih_l  = b_ih + (size_t)l * G3;
    const float*          bhh_l  = b_hh + (size_t)l * G3;

    for (int t0 = 0; t0 < T_DIM; t0 += lag) {
      gru_step<<<dim3(D_DIM / 128, B_DIM / 32, lag), 256, 0, stream>>>(
          in_bf, out_bf, out_f, wih_l, whh_l, bih_l, bhh_l,
          ring_f, ring_b, t0, lag, ringsz);
    }
  }
}
